// LSTMEncoder_25022479466541
// MI455X (gfx1250) — matrix operand order repaired and checked by reading, not compiled, not run on hardware
//
#include <hip/hip_runtime.h>
#include <cstddef>
#include <cstdint>

// ---------------- types ----------------
typedef __attribute__((ext_vector_type(16))) __bf16 bf16x16;
typedef __attribute__((ext_vector_type(8)))  __bf16 bf16x8;
typedef __attribute__((ext_vector_type(8)))  float  v8f;
typedef __attribute__((ext_vector_type(4)))  unsigned int u32x4;
typedef __attribute__((ext_vector_type(8)))  int i32x8;
typedef __attribute__((ext_vector_type(4)))  int i32x4;

static constexpr int BATCH = 64;
static constexpr int TLEN  = 1024;
static constexpr int HID_  = 1024;
static constexpr int FOURH = 4096;

// ---------------- conversion kernels ----------------
__global__ void cvt_f32_to_bf16(const float* __restrict__ src,
                                __bf16* __restrict__ dst, long long n) {
    long long i = (long long)blockIdx.x * blockDim.x + threadIdx.x;
    if (i < n) dst[i] = (__bf16)src[i];
}

// Build Wc[4H, K] bf16 where Wc[j,k] = (k < D) ? Wx[k,j] : Wh[k-D,j]
__global__ void build_wc(const float* __restrict__ Wx, const float* __restrict__ Wh,
                         __bf16* __restrict__ Wc, int D, int K) {
    long long i = (long long)blockIdx.x * blockDim.x + threadIdx.x;
    long long n = (long long)FOURH * K;
    if (i >= n) return;
    int j = (int)(i / K);
    int k = (int)(i % K);
    float v = (k < D) ? Wx[(long long)k * FOURH + j]
                      : Wh[(long long)(k - D) * FOURH + j];
    Wc[i] = (__bf16)v;
}

// ---------------- TDM helper ----------------
// Issue a 2D tensor-tile DMA (global -> LDS): `rows` rows of `width_elems`
// bf16 elements, global row stride `row_stride_elems`, packed tight in LDS.
// D# layout per CDNA5 ISA 8.3 (group0) / 8.4 (group1); groups 2/3 unused (2D).
__device__ __forceinline__ unsigned lds_offset_of(const void* p) {
    // Generic pointers to LDS carry the byte offset in the low 32 bits.
    return (unsigned)(uintptr_t)p;
}

__device__ __forceinline__ void tdm_load_tile_2d(
    unsigned lds_off_bytes, const void* gptr,
    unsigned width_elems, unsigned rows, unsigned long long row_stride_elems)
{
    unsigned long long ga = (unsigned long long)(uintptr_t)gptr;
    u32x4 g0;
    g0[0] = 1u;                                            // count=1, user desc
    g0[1] = lds_off_bytes;                                 // lds_addr
    g0[2] = (unsigned)ga;                                  // global_addr[31:0]
    g0[3] = (unsigned)((ga >> 32) & 0x01ffffffu)           // global_addr[56:32]
            | (2u << 30);                                  // type=2 ("image")
    i32x8 g1;
    g1[0] = 0x00010000;                                    // data_size=1 (2 bytes)
    g1[1] = (int)((width_elems & 0xffffu) << 16);          // tensor_dim0[15:0]
    g1[2] = (int)(((width_elems >> 16) & 0xffffu)          // tensor_dim0[31:16]
            | ((rows & 0xffffu) << 16));                   // tensor_dim1[15:0]
    g1[3] = (int)(((rows >> 16) & 0xffffu)                 // tensor_dim1[31:16]
            | ((width_elems & 0xffffu) << 16));            // tile_dim0
    g1[4] = (int)(rows & 0xffffu);                         // tile_dim1 (tile_dim2=0)
    g1[5] = (int)(row_stride_elems & 0xffffffffull);       // dim0_stride[31:0]
    g1[6] = (int)((row_stride_elems >> 32) & 0xffffull);   // dim0_stride[47:32]
    g1[7] = 0;
    i32x4 z4 = {0, 0, 0, 0};
    i32x8 z8 = {0, 0, 0, 0, 0, 0, 0, 0};
    // 6-arg form (clang-23 / therock-10.0 headers): groups 0..3 + extra group + cpol.
    __builtin_amdgcn_tensor_load_to_lds(g0, g1, z4, z4, z8, 0);
}

// ---------------- fused LSTM timestep ----------------
// Grid: (64 n-tiles, 4 batch-row-groups). Block: 128 threads = 4 waves,
// wave g = gate g (i,f,g,o). Each block TDM-stages its 16-row A panel
// (x slice [16 x D] + h slice [16 x H]) into LDS once, then runs two
// barrier-free affine K loops of v_wmma_f32_16x16x32_bf16, then fuses gates.
__device__ __forceinline__ float sigmoidf_fast(float x) {
    return 1.0f / (1.0f + __expf(-x));
}
__device__ __forceinline__ float tanhf_fast(float x) {
    float e = __expf(2.0f * x);
    return (e - 1.0f) / (e + 1.0f);
}

template <int D, int KTOT>
__global__ __launch_bounds__(128) void lstm_step(
    const __bf16* __restrict__ xsrc,   // [B, T, D] bf16 (prev layer output or xs)
    const __bf16* __restrict__ Wc,     // [4H, KTOT] bf16, K-contiguous
    const float*  __restrict__ bias,   // [4H] f32
    float*        __restrict__ cbuf,   // [B, H] f32 cell state
    __bf16*       __restrict__ hbuf,   // [B, T, H] bf16: read t-1, write t
    float*        __restrict__ fout,   // optional f32 [B, T, H] (final layer)
    int t)
{
    __shared__ __align__(16) __bf16 Ax[16 * D];      // x panel (16-32 KB)
    __shared__ __align__(16) __bf16 Ah[16 * HID_];   // h panel (32 KB)
    __shared__ float zs[4 * 16 * 16];                // i,f,g,o 16x16 z tiles (4 KB)

    const int tid  = threadIdx.x;
    const int wave = tid >> 5;       // gate index 0..3
    const int lane = tid & 31;
    const int n0   = blockIdx.x * 16;
    const int b0   = blockIdx.y * 16;
    const int j0   = wave * HID_ + n0;

    // Stage A panel via Tensor Data Mover (wave 0 issues; TENSORcnt + barrier).
    if (wave == 0) {
        tdm_load_tile_2d(lds_offset_of(Ax),
                         xsrc + ((size_t)b0 * TLEN + t) * D,
                         (unsigned)D, 16u, (unsigned long long)TLEN * D);
        if (t > 0) {
            tdm_load_tile_2d(lds_offset_of(Ah),
                             hbuf + ((size_t)b0 * TLEN + (t - 1)) * HID_,
                             (unsigned)HID_, 16u, (unsigned long long)TLEN * HID_);
        }
        __builtin_amdgcn_s_wait_tensorcnt(0);
    }
    __syncthreads();

    const int col   = lane & 15;
    const int khalf = lane >> 4;     // 0 or 1
    const int cb    = khalf * 8;

    // Per-lane affine base pointers (no per-iteration selects).
    const __bf16* axrow = Ax + col * D + khalf * 16;
    const __bf16* ahrow = Ah + col * HID_ + khalf * 16;
    const __bf16* wrow  = Wc + (size_t)(j0 + col) * KTOT + khalf * 16;

    v8f acc = v8f{};

    // x region: K = 0 .. D
    for (int k0 = 0; k0 < D; k0 += 32) {
        bf16x8 alo = *(const bf16x8*)(axrow + k0);
        bf16x8 ahi = *(const bf16x8*)(axrow + k0 + 8);
        bf16x16 amat;
        #pragma unroll
        for (int i = 0; i < 8; ++i) { amat[i] = alo[i]; amat[8 + i] = ahi[i]; }
        bf16x16 bmat = *(const bf16x16*)(wrow + k0);
        acc = __builtin_amdgcn_wmma_f32_16x16x32_bf16(
            false, amat, false, bmat, (short)0, acc, false, false);
    }
    // h region: K = D .. D+H (empty at t==0 since h == 0)
    const int hlim = (t > 0) ? HID_ : 0;
    for (int k0 = 0; k0 < hlim; k0 += 32) {
        bf16x8 alo = *(const bf16x8*)(ahrow + k0);
        bf16x8 ahi = *(const bf16x8*)(ahrow + k0 + 8);
        bf16x16 amat;
        #pragma unroll
        for (int i = 0; i < 8; ++i) { amat[i] = alo[i]; amat[8 + i] = ahi[i]; }
        bf16x16 bmat = *(const bf16x16*)(wrow + D + k0);
        acc = __builtin_amdgcn_wmma_f32_16x16x32_bf16(
            false, amat, false, bmat, (short)0, acc, false, false);
    }

    // Spill z tile: VGPR v -> local row khalf*8+v, col.
    #pragma unroll
    for (int v = 0; v < 8; ++v)
        zs[(wave * 16 + cb + v) * 16 + col] = acc[v];
    __syncthreads();

    // Fused gates: 16x16 = 256 outputs, 2 per thread.
    #pragma unroll
    for (int q = 0; q < 2; ++q) {
        int e  = tid * 2 + q;
        int bl = e >> 4;
        int n  = e & 15;
        int b  = b0 + bl;
        float zi = zs[(0 * 16 + bl) * 16 + n] + bias[0 * HID_ + n0 + n];
        float zf = zs[(1 * 16 + bl) * 16 + n] + bias[1 * HID_ + n0 + n];
        float zg = zs[(2 * 16 + bl) * 16 + n] + bias[2 * HID_ + n0 + n];
        float zo = zs[(3 * 16 + bl) * 16 + n] + bias[3 * HID_ + n0 + n];
        float gi = sigmoidf_fast(zi);
        float gf = sigmoidf_fast(zf);
        float gg = tanhf_fast(zg);
        float go = sigmoidf_fast(zo);
        size_t cidx = (size_t)b * HID_ + n0 + n;
        float cp = (t > 0) ? cbuf[cidx] : 0.0f;
        float cn = gf * cp + gi * gg;
        float h  = go * tanhf_fast(cn);
        cbuf[cidx] = cn;
        size_t hidx = ((size_t)b * TLEN + t) * HID_ + n0 + n;
        hbuf[hidx] = (__bf16)h;
        if (fout) fout[hidx] = h;
    }
}

// ---------------- host launcher ----------------
extern "C" void kernel_launch(void* const* d_in, const int* in_sizes, int n_in,
                              void* d_out, int out_size, void* d_ws, size_t ws_size,
                              hipStream_t stream) {
    (void)in_sizes; (void)n_in; (void)out_size; (void)ws_size;
    const float* xs  = (const float*)d_in[0];
    const float* Wx0 = (const float*)d_in[1];
    const float* Wh0 = (const float*)d_in[2];
    const float* b0  = (const float*)d_in[3];
    const float* Wx1 = (const float*)d_in[4];
    const float* Wh1 = (const float*)d_in[5];
    const float* b1  = (const float*)d_in[6];
    const float* Wx2 = (const float*)d_in[7];
    const float* Wh2 = (const float*)d_in[8];
    const float* b2  = (const float*)d_in[9];

    // Workspace carve-out (all offsets 256B aligned).
    char* p = (char*)d_ws;
    auto carve = [&](size_t bytes) {
        void* r = (void*)p;
        p += (bytes + 255) & ~(size_t)255;
        return r;
    };
    const size_t FEAT = 512;
    __bf16* xbf = (__bf16*)carve((size_t)BATCH * TLEN * FEAT * 2);   // 64 MB
    __bf16* h0  = (__bf16*)carve((size_t)BATCH * TLEN * HID_ * 2);   // 128 MB
    __bf16* h1  = (__bf16*)carve((size_t)BATCH * TLEN * HID_ * 2);   // 128 MB
    __bf16* Wc0 = (__bf16*)carve((size_t)FOURH * 1536 * 2);          // 12 MB
    __bf16* Wc1 = (__bf16*)carve((size_t)FOURH * 2048 * 2);          // 16 MB
    __bf16* Wc2 = (__bf16*)carve((size_t)FOURH * 2048 * 2);          // 16 MB
    float*  cb  = (float*)carve((size_t)BATCH * HID_ * 4);           // 256 KB

    // 1) Convert inputs / build transposed bf16 weights.
    {
        long long n = (long long)BATCH * TLEN * FEAT;
        cvt_f32_to_bf16<<<(unsigned)((n + 255) / 256), 256, 0, stream>>>(xs, xbf, n);
    }
    {
        long long n = (long long)FOURH * 1536;
        build_wc<<<(unsigned)((n + 255) / 256), 256, 0, stream>>>(Wx0, Wh0, Wc0, 512, 1536);
    }
    {
        long long n = (long long)FOURH * 2048;
        build_wc<<<(unsigned)((n + 255) / 256), 256, 0, stream>>>(Wx1, Wh1, Wc1, 1024, 2048);
        build_wc<<<(unsigned)((n + 255) / 256), 256, 0, stream>>>(Wx2, Wh2, Wc2, 1024, 2048);
    }

    dim3 grid(HID_ / 16, BATCH / 16);   // 64 column tiles x 4 batch row groups
    dim3 block(128);                    // 4 waves: one per gate

    // 2) Layer 0: xs(bf16) -> h0
    for (int t = 0; t < TLEN; ++t)
        lstm_step<512, 1536><<<grid, block, 0, stream>>>(xbf, Wc0, b0, cb, h0, nullptr, t);
    // 3) Layer 1: h0 -> h1
    for (int t = 0; t < TLEN; ++t)
        lstm_step<1024, 2048><<<grid, block, 0, stream>>>(h0, Wc1, b1, cb, h1, nullptr, t);
    // 4) Layer 2: h1 -> (h0 reused as bf16 state) + f32 d_out
    for (int t = 0; t < TLEN; ++t)
        lstm_step<1024, 2048><<<grid, block, 0, stream>>>(h1, Wc2, b2, cb, h0,
                                                          (float*)d_out, t);
}
